// UnfoldLayer_15556371546960
// MI455X (gfx1250) — matrix operand order repaired and checked by reading, not compiled, not run on hardware
//
#include <hip/hip_runtime.h>
#include <hip/hip_bf16.h>
#include <stdint.h>

// ---- problem dims ----
#define BB     8
#define DD     1024
#define TOPC   256
#define CIN    64
#define DOWN   128
#define BR     2
#define NR     8
#define GG     8
#define HH     64
#define WW     64
#define HID    512
#define DBRNR  (DD*BR*NR)     // 16384 sample points per batch
#define ROWS_X (BB*DD)        // 8192  rows of x
#define ROWS_S (BB*DBRNR)     // 131072 rows of S
#define ROWS_O (BB*DD*BR)     // 16384 output rows

typedef __attribute__((ext_vector_type(16))) __bf16 v16bf;
typedef __attribute__((ext_vector_type(8)))  float  v8f;

__device__ __forceinline__ uint16_t f2bf(float f) {
    uint32_t u = __float_as_uint(f);
    uint32_t r = u + 0x7FFFu + ((u >> 16) & 1u);
    return (uint16_t)(r >> 16);
}

union FragU { uint4 u[2]; v16bf v; };

// A fragment (16x32 bf16, row-major MxK source): lane row = M, split K chunks
__device__ __forceinline__ v16bf load_fragA(const uint16_t* base, int row, int K, int kofs, int half) {
    const uint16_t* p = base + (size_t)row * K + kofs + half * 8;
    FragU f;
    f.u[0] = *(const uint4*)(p);        // K = kofs + half*8 + 0..7
    f.u[1] = *(const uint4*)(p + 16);   // K = kofs + 16 + half*8 + 0..7
    return f.v;
}

// B fragment (32x16 bf16, weights stored NxK row-major): lane row = N, split K chunks
__device__ __forceinline__ v16bf load_fragB(const uint16_t* base, int row, int K, int kofs, int half) {
    const uint16_t* p = base + (size_t)row * K + kofs + half * 8;
    FragU f;
    f.u[0] = *(const uint4*)(p);        // K = kofs + half*8 + 0..7
    f.u[1] = *(const uint4*)(p + 16);   // K = kofs + 16 + half*8 + 0..7
    return f.v;
}

__device__ __forceinline__ float gelu_exact(float x) {
    return 0.5f * x * (1.0f + erff(x * 0.70710678118654752f));
}

// ---------------- kernel 0: f32 -> bf16 convert ----------------
__global__ void k_f2bf(const float* __restrict__ in, uint16_t* __restrict__ out, int n) {
    int i = blockIdx.x * blockDim.x + threadIdx.x;
    if (i < n) out[i] = f2bf(in[i]);
}

// ---------------- kernel 1: (B,C,H,W) -> (B,H,W,C) transpose for coalesced gathers ----------------
__global__ void k_transpose(const float* __restrict__ I, const float* __restrict__ IX,
                            const float* __restrict__ IY, const float* __restrict__ IT,
                            float* __restrict__ dst) {
    __shared__ float lds[64][64];                 // skewed, conflict-free
    int bid = blockIdx.x;                         // 4 * B * H = 2048
    int arr = bid >> 9;
    int rem = bid & 511;
    int b = rem >> 6;
    int xr = rem & 63;
    const float* src = (arr == 0) ? I : (arr == 1) ? IX : (arr == 2) ? IY : IT;
    const float* sp = src + (((size_t)b * CIN) * HH + xr) * WW;   // + c*H*W + y
    int t = threadIdx.x;
    int y = t & 63, c0 = t >> 6;
    for (int i = 0; i < 16; i++) {
        int c = c0 + i * 4;
        lds[c][(y + c) & 63] = sp[(size_t)c * HH * WW + y];
    }
    __syncthreads();
    float* dp = dst + (size_t)arr * ((size_t)BB * HH * WW * CIN)
                    + (((size_t)b * HH + xr) * WW) * CIN;         // + y*C + c
    int c = t & 63, y0 = t >> 6;
    for (int i = 0; i < 16; i++) {
        int yy = y0 + i * 4;
        dp[(size_t)yy * CIN + c] = lds[c][(yy + c) & 63];
    }
}

// ---------------- kernel 2: GEMM1  x(8192x256) x {w_idx(64x256), w_mask(128x256)}^T ----------------
// Register-blocked: A fragments (K=256 -> 8 frags) loaded once, all 12 N-tiles per wave.
__global__ void k_gemm1(const uint16_t* __restrict__ xbf, const uint16_t* __restrict__ wibf,
                        const uint16_t* __restrict__ wmbf, const float* __restrict__ b_idx,
                        const float* __restrict__ b_mask, float* __restrict__ p_ws,
                        float* __restrict__ maskws) {
    int mt = blockIdx.x;
    int lane = threadIdx.x;
    int nl = lane & 15, half = lane >> 4;
    int rowA = mt * 16 + nl;
    v16bf a[8];
    for (int ks = 0; ks < 8; ks++) a[ks] = load_fragA(xbf, rowA, TOPC, ks * 32, half);
    for (int nt = 0; nt < 12; nt++) {
        const uint16_t* wb;
        int rowB;
        if (nt < 4) { wb = wibf; rowB = nt * 16 + nl; }
        else        { wb = wmbf; rowB = (nt - 4) * 16 + nl; }
        v8f c = {};
        for (int ks = 0; ks < 8; ks++) {
            v16bf b = load_fragB(wb, rowB, TOPC, ks * 32, half);
            c = __builtin_amdgcn_wmma_f32_16x16x32_bf16(false, a[ks], false, b, (short)0, c, false, false);
        }
        if (nt < 4) {
            int j = nt * 16 + nl;                       // 0..63 -> (br, nr, k)
            int k = j & 3, q = j >> 2, br = q >> 3, nr = q & 7;
            float bias = b_idx[j];
            for (int v = 0; v < 8; v++) {
                int m = mt * 16 + v + 8 * half;
                int bb = m >> 10, d = m & 1023;
                int n = (d * 2 + br) * 8 + nr;
                float val = c[v] + bias;
                float s = 1.0f / (1.0f + __expf(-val));
                p_ws[(((size_t)bb * DBRNR + n) << 2) + k] = s * 63.0f;   // fac = H-1 = W-1 = 63
            }
        } else {
            int j = (nt - 4) * 16 + nl;                 // 0..127 -> (br, nr, g)
            int br = j >> 6, nr = (j >> 3) & 7, g = j & 7;
            float bias = b_mask[j];
            for (int v = 0; v < 8; v++) {
                int m = mt * 16 + v + 8 * half;
                int bb = m >> 10, d = m & 1023;
                size_t orow = ((size_t)bb * DD + d) * 2 + br;
                maskws[(orow * GG + g) * NR + nr] = c[v] + bias;   // nr contiguous for softmax
            }
        }
    }
}

// ---------------- kernel 3: softmax over NR=8 ----------------
__global__ void k_softmax8(float* __restrict__ m) {
    int r = blockIdx.x * blockDim.x + threadIdx.x;
    if (r >= ROWS_O * GG) return;
    float* p = m + (size_t)r * NR;
    float v[NR], vmax = -1e30f;
    for (int i = 0; i < NR; i++) { v[i] = p[i]; vmax = fmaxf(vmax, v[i]); }
    float s = 0.f;
    for (int i = 0; i < NR; i++) { v[i] = __expf(v[i] - vmax); s += v[i]; }
    float inv = 1.0f / s;
    for (int i = 0; i < NR; i++) p[i] = v[i] * inv;
}

// ---------------- kernel 4: box sampling (coalesced over channels) ----------------
__device__ __forceinline__ float sample_corner(const float* __restrict__ PI,
                                               const float* __restrict__ PIX,
                                               const float* __restrict__ PIY,
                                               const float* __restrict__ PIT,
                                               int b, int c, float xc, float yc) {
    int X = (int)ceilf(xc), Y = (int)ceilf(yc);
    float dx = (float)X - xc, dy = (float)Y - yc;
    float wx1 = 0.5f * dx * dx, wx2 = dx - wx1;
    float wy1 = 0.5f * dy * dy, wy2 = dy - wy1;
#define AT(P, xi, yi) (P)[((((size_t)b * HH + (xi)) * WW) + (yi)) * CIN + c]
    float s_a  = AT(PIT, X, Y);
    float s_e1 = wy1 * (AT(PIX, X, Y-1) - 0.5f * AT(PI, X, Y-1) - 0.5f * AT(PI, 0, Y-1))
               + wy2 * (AT(PIX, X, Y)   - 0.5f * AT(PI, X, Y)   - 0.5f * AT(PI, 0, Y));
    float s_e2 = wx1 * (AT(PIY, X-1, Y) - 0.5f * AT(PI, X-1, Y) - 0.5f * AT(PI, X-1, 0))
               + wx2 * (AT(PIY, X, Y)   - 0.5f * AT(PI, X, Y)   - 0.5f * AT(PI, X, 0));
    float s_c  = wx1 * wy1 * AT(PI, X-1, Y-1) + wx2 * wy1 * AT(PI, X, Y-1)
               + wx1 * wy2 * AT(PI, X-1, Y)   + wx2 * wy2 * AT(PI, X, Y);
#undef AT
    return s_a - s_e1 - s_e2 + s_c;
}

__global__ void k_sample(const float* __restrict__ imgT, const float* __restrict__ p_ws,
                         uint16_t* __restrict__ Sbf) {
    int row = blockIdx.x;              // b*16384 + n
    int b = row >> 14;
    int c = threadIdx.x;               // channel, 64 threads
    const float4 p4 = ((const float4*)p_ws)[row];
    float x1 = p4.x, y1 = p4.y, x2 = p4.z, y2 = p4.w;
    const size_t AS = (size_t)BB * HH * WW * CIN;
    const float* PI  = imgT;
    const float* PIX = imgT + AS;
    const float* PIY = imgT + 2 * AS;
    const float* PIT = imgT + 3 * AS;
    float acc = sample_corner(PI, PIX, PIY, PIT, b, c, x2, y2)
              - sample_corner(PI, PIX, PIY, PIT, b, c, x1, y2)
              - sample_corner(PI, PIX, PIY, PIT, b, c, x2, y1)
              + sample_corner(PI, PIX, PIY, PIT, b, c, x1, y1);
    Sbf[(size_t)row * CIN + c] = f2bf(acc);
}

// ---------------- kernel 5: GEMM2 rects = gelu(S @ w_proj^T + b), mask-weighted NR-reduce ----------------
// C-layout trick: lanes 0..15 hold tile rows 0..7 (= NR group 0) in their 8 acc VGPRs,
// lanes 16..31 hold rows 8..15 (= NR group 1) -> sum over NR is an in-lane reduction.
__global__ void k_gemm2(const uint16_t* __restrict__ Sbf, const uint16_t* __restrict__ wpbf,
                        const float* __restrict__ b_proj, const float* __restrict__ maskws,
                        float* __restrict__ y1raw) {
    int t = blockIdx.x;                    // 8192 tiles of 16 S-rows = 2 output rows
    int lane = threadIdx.x, nl = lane & 15, half = lane >> 4;
    int rowA = t * 16 + nl;
    v16bf a0 = load_fragA(Sbf, rowA, CIN, 0, half);
    v16bf a1 = load_fragA(Sbf, rowA, CIN, 32, half);
    size_t orow = (size_t)t * 2 + half;
    for (int ntile = 0; ntile < 8; ntile++) {
        int rowB = ntile * 16 + nl;
        v16bf b0 = load_fragB(wpbf, rowB, CIN, 0, half);
        v16bf b1 = load_fragB(wpbf, rowB, CIN, 32, half);
        v8f cc = {};
        cc = __builtin_amdgcn_wmma_f32_16x16x32_bf16(false, a0, false, b0, (short)0, cc, false, false);
        cc = __builtin_amdgcn_wmma_f32_16x16x32_bf16(false, a1, false, b1, (short)0, cc, false, false);
        int col = ntile * 16 + nl;
        int g = col >> 4;
        float bias = b_proj[col];
        const float* mp = maskws + (orow * GG + g) * NR;
        float acc = 0.f;
        for (int v = 0; v < 8; v++) {
            float x = cc[v] + bias;           // v == nr within the group
            acc += mp[v] * gelu_exact(x);
        }
        y1raw[orow * DOWN + col] = acc;
    }
}

// ---------------- kernel 6: LayerNorm(128) -> f32 shortcut + bf16 activation ----------------
__global__ void k_ln1(const float* __restrict__ in, const float* __restrict__ g,
                      const float* __restrict__ be, float* __restrict__ outf,
                      uint16_t* __restrict__ outbf) {
    __shared__ float2 red[DOWN];
    int r = blockIdx.x, t = threadIdx.x;
    float v = in[(size_t)r * DOWN + t];
    red[t] = make_float2(v, v * v);
    __syncthreads();
    for (int s = DOWN / 2; s > 0; s >>= 1) {
        if (t < s) { red[t].x += red[t + s].x; red[t].y += red[t + s].y; }
        __syncthreads();
    }
    float mu  = red[0].x * (1.0f / DOWN);
    float var = red[0].y * (1.0f / DOWN) - mu * mu;
    float o = (v - mu) * rsqrtf(var + 1e-5f) * g[t] + be[t];
    outf[(size_t)r * DOWN + t]  = o;
    outbf[(size_t)r * DOWN + t] = f2bf(o);
}

// ---------------- kernel 7: MLP1 (16384x128)x(512x128)^T + gelu -> bf16 ----------------
// Register-blocked: A frags (K=128 -> 4) loaded once; N processed in 4 groups of 8 tiles.
__global__ void k_gemm3(const uint16_t* __restrict__ A, const uint16_t* __restrict__ W,
                        const float* __restrict__ bias, uint16_t* __restrict__ Hout) {
    int mt = blockIdx.x;
    int lane = threadIdx.x, nl = lane & 15, half = lane >> 4;
    int rowA = mt * 16 + nl;
    v16bf a[4];
    for (int ks = 0; ks < 4; ks++) a[ks] = load_fragA(A, rowA, DOWN, ks * 32, half);
    for (int ng = 0; ng < 4; ng++) {
        v8f acc[8] = {};
        for (int ks = 0; ks < 4; ks++) {
            for (int j = 0; j < 8; j++) {
                int rowB = (ng * 8 + j) * 16 + nl;
                v16bf b = load_fragB(W, rowB, DOWN, ks * 32, half);
                acc[j] = __builtin_amdgcn_wmma_f32_16x16x32_bf16(false, a[ks], false, b, (short)0, acc[j], false, false);
            }
        }
        for (int j = 0; j < 8; j++) {
            int col = (ng * 8 + j) * 16 + nl;
            float bi = bias[col];
            for (int v = 0; v < 8; v++) {
                float x = acc[j][v] + bi;
                Hout[((size_t)(mt * 16 + v + 8 * half)) * HID + col] = f2bf(gelu_exact(x));
            }
        }
    }
}

// ---------------- kernel 8: MLP2 (16384x512)x(128x512)^T -> f32 ----------------
// K-outer loop: each A frag loaded once, accumulated into all 8 N-tiles.
__global__ void k_gemm4(const uint16_t* __restrict__ A, const uint16_t* __restrict__ W,
                        const float* __restrict__ bias, float* __restrict__ out) {
    int mt = blockIdx.x;
    int lane = threadIdx.x, nl = lane & 15, half = lane >> 4;
    int rowA = mt * 16 + nl;
    v8f acc[8] = {};
    for (int ks = 0; ks < 16; ks++) {
        v16bf a = load_fragA(A, rowA, HID, ks * 32, half);
        for (int nt = 0; nt < 8; nt++) {
            v16bf b = load_fragB(W, nt * 16 + nl, HID, ks * 32, half);
            acc[nt] = __builtin_amdgcn_wmma_f32_16x16x32_bf16(false, a, false, b, (short)0, acc[nt], false, false);
        }
    }
    for (int nt = 0; nt < 8; nt++) {
        int col = nt * 16 + nl;
        float bi = bias[col];
        for (int v = 0; v < 8; v++)
            out[((size_t)(mt * 16 + v + 8 * half)) * DOWN + col] = acc[nt][v] + bi;
    }
}

// ---------------- kernel 9: LayerNorm + residual -> d_out ----------------
__global__ void k_ln2(const float* __restrict__ in, const float* __restrict__ g,
                      const float* __restrict__ be, const float* __restrict__ shortcut,
                      float* __restrict__ out) {
    __shared__ float2 red[DOWN];
    int r = blockIdx.x, t = threadIdx.x;
    float v = in[(size_t)r * DOWN + t];
    red[t] = make_float2(v, v * v);
    __syncthreads();
    for (int s = DOWN / 2; s > 0; s >>= 1) {
        if (t < s) { red[t].x += red[t + s].x; red[t].y += red[t + s].y; }
        __syncthreads();
    }
    float mu  = red[0].x * (1.0f / DOWN);
    float var = red[0].y * (1.0f / DOWN) - mu * mu;
    float o = (v - mu) * rsqrtf(var + 1e-5f) * g[t] + be[t];
    out[(size_t)r * DOWN + t] = o + shortcut[(size_t)r * DOWN + t];
}

extern "C" void kernel_launch(void* const* d_in, const int* in_sizes, int n_in,
                              void* d_out, int out_size, void* d_ws, size_t ws_size,
                              hipStream_t stream) {
    (void)in_sizes; (void)n_in; (void)out_size; (void)ws_size;
    const float* x      = (const float*)d_in[0];
    const float* I      = (const float*)d_in[1];
    const float* IX     = (const float*)d_in[2];
    const float* IY     = (const float*)d_in[3];
    const float* IT     = (const float*)d_in[4];
    const float* w_idx  = (const float*)d_in[5];
    const float* b_idx  = (const float*)d_in[6];
    const float* w_proj = (const float*)d_in[7];
    const float* b_proj = (const float*)d_in[8];
    const float* w_mask = (const float*)d_in[9];
    const float* b_mask = (const float*)d_in[10];
    const float* g1     = (const float*)d_in[11];
    const float* beta1  = (const float*)d_in[12];
    const float* w_mlp1 = (const float*)d_in[13];
    const float* b_mlp1 = (const float*)d_in[14];
    const float* w_mlp2 = (const float*)d_in[15];
    const float* b_mlp2 = (const float*)d_in[16];
    const float* g2     = (const float*)d_in[17];
    const float* beta2  = (const float*)d_in[18];

    char* w = (char*)d_ws;
    size_t off = 0;
    auto alloc = [&](size_t bytes) -> void* {
        void* p = w + off;
        off = (off + bytes + 255) & ~(size_t)255;
        return p;
    };
    float*    imgT   = (float*)   alloc((size_t)4 * BB * HH * WW * CIN * 4);   // 32 MB
    uint16_t* xbf    = (uint16_t*)alloc((size_t)ROWS_X * TOPC * 2);
    uint16_t* wibf   = (uint16_t*)alloc((size_t)64 * TOPC * 2);
    uint16_t* wmbf   = (uint16_t*)alloc((size_t)128 * TOPC * 2);
    uint16_t* wpbf   = (uint16_t*)alloc((size_t)128 * CIN * 2);
    uint16_t* w1bf   = (uint16_t*)alloc((size_t)HID * DOWN * 2);
    uint16_t* w2bf   = (uint16_t*)alloc((size_t)DOWN * HID * 2);
    float*    p_ws   = (float*)   alloc((size_t)ROWS_S * 4 * 4);
    float*    maskws = (float*)   alloc((size_t)ROWS_O * GG * NR * 4);
    uint16_t* Sbf    = (uint16_t*)alloc((size_t)ROWS_S * CIN * 2);
    float*    y1raw  = (float*)   alloc((size_t)ROWS_O * DOWN * 4);
    float*    ln1f   = (float*)   alloc((size_t)ROWS_O * DOWN * 4);
    uint16_t* ln1bf  = (uint16_t*)alloc((size_t)ROWS_O * DOWN * 2);
    uint16_t* hidbf  = (uint16_t*)alloc((size_t)ROWS_O * HID * 2);
    float*    y2raw  = (float*)   alloc((size_t)ROWS_O * DOWN * 4);

    // 0: bf16 conversions
    auto cvt = [&](const float* src, uint16_t* dst, int n) {
        k_f2bf<<<(n + 255) / 256, 256, 0, stream>>>(src, dst, n);
    };
    cvt(x,      xbf,  ROWS_X * TOPC);
    cvt(w_idx,  wibf, 64 * TOPC);
    cvt(w_mask, wmbf, 128 * TOPC);
    cvt(w_proj, wpbf, 128 * CIN);
    cvt(w_mlp1, w1bf, HID * DOWN);
    cvt(w_mlp2, w2bf, DOWN * HID);

    // 1: image layout transpose (B,C,H,W) -> (B,H,W,C)
    k_transpose<<<4 * BB * HH, 256, 0, stream>>>(I, IX, IY, IT, imgT);

    // 2: fused idx+mask GEMM (WMMA bf16, A register-resident)
    k_gemm1<<<ROWS_X / 16, 32, 0, stream>>>(xbf, wibf, wmbf, b_idx, b_mask, p_ws, maskws);

    // 3: softmax over NR
    k_softmax8<<<(ROWS_O * GG + 255) / 256, 256, 0, stream>>>(maskws);

    // 4: box sampling
    k_sample<<<ROWS_S, CIN, 0, stream>>>(imgT, p_ws, Sbf);

    // 5: rects GEMM + gelu + masked NR-reduce (WMMA bf16)
    k_gemm2<<<ROWS_S / 16, 32, 0, stream>>>(Sbf, wpbf, b_proj, maskws, y1raw);

    // 6: LN1
    k_ln1<<<ROWS_O, DOWN, 0, stream>>>(y1raw, g1, beta1, ln1f, ln1bf);

    // 7: MLP1 + gelu (WMMA bf16, A register-resident)
    k_gemm3<<<ROWS_O / 16, 32, 0, stream>>>(ln1bf, w1bf, b_mlp1, hidbf);

    // 8: MLP2 (WMMA bf16, K-outer, A loaded once)
    k_gemm4<<<ROWS_O / 16, 32, 0, stream>>>(hidbf, w2bf, b_mlp2, y2raw);

    // 9: LN2 + residual
    k_ln2<<<ROWS_O, DOWN, 0, stream>>>(y2raw, g2, beta2, ln1f, (float*)d_out);
}
